// Model_31988916420723
// MI455X (gfx1250) — matrix operand order repaired and checked by reading, not compiled, not run on hardware
//
#include <hip/hip_runtime.h>
#include <hip/hip_bf16.h>

typedef __attribute__((ext_vector_type(16))) _Float16 v16h;
typedef __attribute__((ext_vector_type(8)))  float    v8f;

#define N_NODES 100000
#define N_EDGES 2500000

// ---------------------------------------------------------------------------
// 0) zero scratch (cnt, agg1, agg2) -- d_ws is poisoned, must init every call
// ---------------------------------------------------------------------------
__global__ void zero_ws_kernel(float* __restrict__ p, int n) {
    int i = blockIdx.x * blockDim.x + threadIdx.x;
    if (i < n) p[i] = 0.0f;
}

// ---------------------------------------------------------------------------
// 1) edge pass for layer 1: scalar feature scatter-add + degree count
// ---------------------------------------------------------------------------
__global__ void edge_agg1_kernel(const long long* __restrict__ ei,
                                 const float* __restrict__ x,
                                 float* __restrict__ agg1,
                                 float* __restrict__ cnt) {
    int e = blockIdx.x * blockDim.x + threadIdx.x;
    if (e >= N_EDGES) return;
    int s = (int)ei[e];
    int d = (int)ei[N_EDGES + e];
    atomicAdd(&agg1[d], x[s]);
    atomicAdd(&cnt[d], 1.0f);
}

// ---------------------------------------------------------------------------
// 2) node pass for layer 1: h1[n][k] = relu(mean1[n]*W1l[k] + b1[k] + x[n]*W1r[k])
//    one thread per (n,k); k = tid%32 -> fully coalesced h1 stores
// ---------------------------------------------------------------------------
__global__ void node_layer1_kernel(const float* __restrict__ x,
                                   const float* __restrict__ agg1,
                                   const float* __restrict__ cnt,
                                   const float* __restrict__ W1l,
                                   const float* __restrict__ b1,
                                   const float* __restrict__ W1r,
                                   float* __restrict__ h1) {
    int idx = blockIdx.x * blockDim.x + threadIdx.x;     // n*32 + k
    if (idx >= N_NODES * 32) return;
    int n = idx >> 5;
    int k = idx & 31;
    float mean = agg1[n] / fmaxf(cnt[n], 1.0f);
    float v = mean * W1l[k] + b1[k] + x[n] * W1r[k];
    h1[idx] = fmaxf(v, 0.0f);
}

// ---------------------------------------------------------------------------
// 3) edge pass for layer 2: 32 lanes per edge. A wave = 1 edge: one coalesced
//    128B gather of h1[src][0..31] and 32 coalesced f32 atomics to agg2[dst].
// ---------------------------------------------------------------------------
__global__ void edge_agg2_kernel(const long long* __restrict__ ei,
                                 const float* __restrict__ h1,
                                 float* __restrict__ agg2) {
    int t = blockIdx.x * blockDim.x + threadIdx.x;       // e*32 + k
    if (t >= N_EDGES * 32) return;
    int e = t >> 5;
    int k = t & 31;
    int s = (int)ei[e];                                  // same addr across wave -> broadcast
    int d = (int)ei[N_EDGES + e];
    atomicAdd(&agg2[d * 32 + k], h1[s * 32 + k]);
}

// ---------------------------------------------------------------------------
// 4) layer-2 SAGEConv + ReLU + output projection, one wave per 16-node tile.
//    h2[16x16] = relu( mean2[16x32] @ W2l^T  +  h1[16x32] @ W2r^T  + b2 )
//    via two chained v_wmma_f32_16x16x32_f16 (K=32 exactly matches F=32),
//    then out[16] = h2 @ W3^T + b3 via LDS staging.
//
//    ISA 7.12.2 operand layouts (wave32):
//      A (16x32 f16):  lane L -> m = L%16, half = L/16;
//                      element i -> K = (i<8) ? half*8+i : 16+half*8+(i-8)
//      B (32x16 f16):  lane L -> n = L%16, half = L/16;
//                      element i -> K as for A; this code loads A and B with K = 16*half + i
//      C/D (16x16 f32): vgpr v -> M = v + 8*half, N = L%16
// ---------------------------------------------------------------------------
__global__ void conv2_out_wmma_kernel(const float* __restrict__ h1,
                                      const float* __restrict__ agg2,
                                      const float* __restrict__ cnt,
                                      const float* __restrict__ W2l,
                                      const float* __restrict__ b2,
                                      const float* __restrict__ W2r,
                                      const float* __restrict__ W3,
                                      const float* __restrict__ b3,
                                      float* __restrict__ out) {
    __shared__ float h2s[16 * 16];                       // one wave per block
    const int tile = blockIdx.x;                         // 6250 tiles of 16 nodes
    const int lane = threadIdx.x;                        // 0..31, EXEC all-1s
    const int half = lane >> 4;
    const int mn   = lane & 15;                          // row for A/C, col for B/C
    const int node = tile * 16 + mn;

    // B operands: B[k][n] = W[n][k]  (row-major [16,32] weights)
    v16h bl, br;
#pragma unroll
    for (int i = 0; i < 16; ++i) {
        int k = 16 * half + i;
        bl[i] = (_Float16)W2l[mn * 32 + k];
        br[i] = (_Float16)W2r[mn * 32 + k];
    }

    // A operands: mean-aggregated tile and root(h1) tile
    const float inv = 1.0f / fmaxf(cnt[node], 1.0f);
    v16h am, ah;
#pragma unroll
    for (int i = 0; i < 16; ++i) {
        int k = 16 * half + i;
        am[i] = (_Float16)(agg2[node * 32 + k] * inv);
        ah[i] = (_Float16)h1[node * 32 + k];
    }

    // D = relu( am*bl + ah*br + b2 )
    v8f c = {};
    c = __builtin_amdgcn_wmma_f32_16x16x32_f16(false, am, false, bl,
                                               (short)0, c, false, false);
    c = __builtin_amdgcn_wmma_f32_16x16x32_f16(false, ah, false, br,
                                               (short)0, c, false, false);

    const float bias = b2[mn];
#pragma unroll
    for (int v = 0; v < 8; ++v) {
        int row = v + 8 * half;
        h2s[row * 16 + mn] = fmaxf(c[v] + bias, 0.0f);
    }
    __syncthreads();                                     // single-wave WG: ~free

    // out[16 nodes] = h2 @ W3^T + b3
    if (lane < 16) {
        float acc = b3[0];
#pragma unroll
        for (int j = 0; j < 16; ++j) acc += h2s[lane * 16 + j] * W3[j];
        out[tile * 16 + lane] = acc;
    }
}

// ---------------------------------------------------------------------------
// launch
// ---------------------------------------------------------------------------
extern "C" void kernel_launch(void* const* d_in, const int* in_sizes, int n_in,
                              void* d_out, int out_size, void* d_ws, size_t ws_size,
                              hipStream_t stream) {
    const float*     x   = (const float*)d_in[0];
    const long long* ei  = (const long long*)d_in[1];   // int64 [2, E]
    // d_in[2] = edge_weight: unused by SAGEConv (matches reference)
    const float* W1l = (const float*)d_in[3];
    const float* b1  = (const float*)d_in[4];
    const float* W1r = (const float*)d_in[5];
    const float* W2l = (const float*)d_in[6];
    const float* b2  = (const float*)d_in[7];
    const float* W2r = (const float*)d_in[8];
    const float* W3  = (const float*)d_in[9];
    const float* b3  = (const float*)d_in[10];
    float* out = (float*)d_out;

    // workspace layout (floats): cnt[N] | agg1[N] | agg2[N*32] | h1[N*32]
    float* ws   = (float*)d_ws;
    float* cnt  = ws;
    float* agg1 = ws + N_NODES;
    float* agg2 = ws + 2 * N_NODES;
    float* h1   = ws + 34 * N_NODES;

    const int nz = 34 * N_NODES;                         // cnt + agg1 + agg2
    zero_ws_kernel<<<(nz + 255) / 256, 256, 0, stream>>>(ws, nz);

    edge_agg1_kernel<<<(N_EDGES + 255) / 256, 256, 0, stream>>>(ei, x, agg1, cnt);

    node_layer1_kernel<<<(N_NODES * 32 + 255) / 256, 256, 0, stream>>>(
        x, agg1, cnt, W1l, b1, W1r, h1);

    edge_agg2_kernel<<<(N_EDGES * 32 + 255) / 256, 256, 0, stream>>>(ei, h1, agg2);

    conv2_out_wmma_kernel<<<N_NODES / 16, 32, 0, stream>>>(
        h1, agg2, cnt, W2l, b2, W2r, W3, b3, out);
}
